// NeuralBridgeSDE_24550033064088
// MI455X (gfx1250) — matrix operand order repaired and checked by reading, not compiled, not run on hardware
//
#include <hip/hip_runtime.h>
#include <math.h>

typedef _Float16 v2h  __attribute__((ext_vector_type(2)));
typedef _Float16 v8h  __attribute__((ext_vector_type(8)));
typedef _Float16 v16h __attribute__((ext_vector_type(16)));
typedef float    v8f  __attribute__((ext_vector_type(8)));

#define BT   16
#define HH   128
#define DX   32
#define DW   32
#define NSTEP 500
#define BB   1024
#define BETA_C 0.5f
#define EPS_C  1e-4f

#if __has_builtin(__builtin_amdgcn_tanhf)
  #define TANHF(x) __builtin_amdgcn_tanhf(x)
#elif __has_builtin(__builtin_amdgcn_tanh_f32)
  #define TANHF(x) __builtin_amdgcn_tanh_f32(x)
#else
  #define TANHF(x) tanhf(x)
#endif
#if __has_builtin(__builtin_amdgcn_rcpf)
  #define RCPF(x) __builtin_amdgcn_rcpf(x)
#else
  #define RCPF(x) (1.f / (x))
#endif
#if __has_builtin(__builtin_amdgcn_sqrtf)
  #define SQRTF(x) __builtin_amdgcn_sqrtf(x)
#else
  #define SQRTF(x) sqrtf(x)
#endif

// Everything transposed: out^T = M^T @ act^T. Constant matrices live as WMMA
// A-fragments in LDS (packed once); activations/noise live [batch][dim] f16 so
// B-fragments and D write-backs are contiguous 16/32-byte LDS accesses.
struct __align__(16) Shared {
    _Float16 Wp1[8 * 512];    // W1[1:,:]^T A-frags (8 mt x 1 kt)
    _Float16 Wp2[32 * 512];   // W2^T      A-frags (8 mt x 4 kt)
    _Float16 Wp3[8 * 512];    // W3^T      A-frags (2 mt x 4 kt)
    _Float16 W3Gf[8 * 512];   // (W3 G^T)^T A-frags (2 mt x 4 kt)  -> vg from h2
    _Float16 Af[2 * 512];     // a = G G^T  A-frags (2 mt x 1 kt)  -> ra from r
    _Float16 Gf[2 * 512];     // G          A-frags (2 mt x 1 kt)  -> wg from dW
    _Float16 xh[BT * DX];     // x^T  f16
    _Float16 rh[BT * DX];     // r^T  f16 (rinv folded in)
    _Float16 dwh[BT * DW];    // dW^T * sqrt(dt) f16
    _Float16 h1t[BT * HH];    // h1^T f16
    _Float16 h2t[BT * HH];    // h2^T f16
    float W1r0[HH];
    float b1v[HH], b2v[HH], b3v[DW], b3g[DW];   // b3g = b3 @ G^T
    float Av[DX * DX];        // a (setup temp for Af pack)
    float muv[DX], vtv[DX];
    float x[BT * DX];         // fp32 SDE state
    float ra[BT * DX];        // r @ a      (WMMA out, fp32)
    float vv[BT * DW];        // v = mlp(x) (fp32, for vs output)
    float vgb[BT * DX];       // v @ G^T
    float wgb[BT * DX];       // dW_scaled @ G^T
    float ll[BT];
};

__device__ inline v8f wmma_f16(v16h a, v16h b, v8f c) {
    return __builtin_amdgcn_wmma_f32_16x16x32_f16(false, a, false, b, (short)0, c, false, false);
}

// B-fragment from [batch][stride] row-major f16 (two 16B reads per lane, table k order).
__device__ inline v16h load_b_act(const _Float16* buf, int stride, int kt, int lane) {
    const int n    = lane & 15;
    const int koff = (lane & 16) ? 8 : 0;
    const v8h lo = *(const v8h*)&buf[n * stride + kt * 32 + koff];
    const v8h hi = *(const v8h*)&buf[n * stride + kt * 32 + 16 + koff];
    return __builtin_shufflevector(lo, hi, 0,1,2,3,4,5,6,7,8,9,10,11,12,13,14,15);
}

__global__ __launch_bounds__(256)
void NeuralBridgeSDE_kernel(const float* __restrict__ x0, const float* __restrict__ ts,
                            const float* __restrict__ dWs, const float* __restrict__ W1,
                            const float* __restrict__ b1, const float* __restrict__ W2,
                            const float* __restrict__ b2, const float* __restrict__ W3,
                            const float* __restrict__ b3, const float* __restrict__ Gm,
                            const float* __restrict__ mu, const float* __restrict__ vtg,
                            float* __restrict__ xs_o, float* __restrict__ vs_o,
                            float* __restrict__ ll_o) {
    __shared__ Shared sh;
    const int tid  = threadIdx.x;
    const int wave = tid >> 5;
    const int lane = tid & 31;
    const int b0   = blockIdx.x * BT;

    const float T     = ts[NSTEP];
    const float rinv0 = RCPF(T - ts[0] + EPS_C);

    // ---------------- one-time setup ----------------
    for (int i = tid; i < HH; i += 256) { sh.b1v[i] = b1[i]; sh.b2v[i] = b2[i]; sh.W1r0[i] = W1[i]; }
    if (tid < DW) sh.b3v[tid] = b3[tid];
    if (tid < DX) { sh.muv[tid] = mu[tid]; sh.vtv[tid] = vtg[tid]; }
    if (tid < DX) {                                   // b3g = b3 @ G^T
        float sgb = 0.f;
        #pragma unroll 4
        for (int k = 0; k < DW; ++k) sgb += b3[k] * Gm[tid * DW + k];
        sh.b3g[tid] = sgb;
    }
    for (int i = tid; i < BT * DX; i += 256) {
        float xv = x0[b0 * DX + i];
        sh.x[i]  = xv;
        sh.xh[i] = (_Float16)xv;
        sh.rh[i] = (_Float16)((vtg[i & 31] - xv) * rinv0);
    }
    if (tid < BT) sh.ll[tid] = 0.f;

    // A-fragment interleave: element e of lane ln holds
    //   M = mt*16 + (ln&15),  K = kt*32 + (e&8?16:0) + (ln&16?8:0) + (e&7).
    for (int idx = tid; idx < 8 * 512; idx += 256) {                 // W1[1:]^T
        int e = idx & 15, ln = (idx >> 4) & 31, mt = (idx >> 9) & 7;
        int K = ((e & 8) ? 16 : 0) + ((ln & 16) ? 8 : 0) + (e & 7);
        sh.Wp1[idx] = (_Float16)W1[(K + 1) * HH + mt * 16 + (ln & 15)];
    }
    for (int idx = tid; idx < 32 * 512; idx += 256) {                // W2^T
        int e = idx & 15, ln = (idx >> 4) & 31, fr = idx >> 9;
        int mt = fr >> 2, kt = fr & 3;
        int K = kt * 32 + ((e & 8) ? 16 : 0) + ((ln & 16) ? 8 : 0) + (e & 7);
        sh.Wp2[idx] = (_Float16)W2[K * HH + mt * 16 + (ln & 15)];
    }
    for (int idx = tid; idx < 8 * 512; idx += 256) {                 // W3^T
        int e = idx & 15, ln = (idx >> 4) & 31, fr = idx >> 9;
        int mt = fr >> 2, kt = fr & 3;
        int K = kt * 32 + ((e & 8) ? 16 : 0) + ((ln & 16) ? 8 : 0) + (e & 7);
        sh.Wp3[idx] = (_Float16)W3[K * DW + mt * 16 + (ln & 15)];
    }
    for (int idx = tid; idx < 8 * 512; idx += 256) {                 // (W3 G^T)^T
        int e = idx & 15, ln = (idx >> 4) & 31, fr = idx >> 9;
        int mt = fr >> 2, kt = fr & 3;
        int K  = kt * 32 + ((e & 8) ? 16 : 0) + ((ln & 16) ? 8 : 0) + (e & 7);
        int Mg = mt * 16 + (ln & 15);
        float sdot = 0.f;
        #pragma unroll 4
        for (int k = 0; k < DW; ++k) sdot += W3[K * DW + k] * Gm[Mg * DW + k];
        sh.W3Gf[idx] = (_Float16)sdot;
    }
    for (int idx = tid; idx < 2 * 512; idx += 256) {                 // G
        int e = idx & 15, ln = (idx >> 4) & 31, mt = idx >> 9;
        int K  = ((e & 8) ? 16 : 0) + ((ln & 16) ? 8 : 0) + (e & 7);
        int Mg = mt * 16 + (ln & 15);
        sh.Gf[idx] = (_Float16)Gm[Mg * DW + K];
    }
    for (int idx = tid; idx < DX * DX; idx += 256) {                 // a = G G^T (fp32 temp)
        int i = idx >> 5, j = idx & 31;
        float sdot = 0.f;
        #pragma unroll 4
        for (int k = 0; k < DW; ++k) sdot += Gm[i * DW + k] * Gm[j * DW + k];
        sh.Av[idx] = sdot;
    }
    __syncthreads();
    for (int idx = tid; idx < 2 * 512; idx += 256) {                 // a -> A-frags
        int e = idx & 15, ln = (idx >> 4) & 31, mt = idx >> 9;
        int K  = ((e & 8) ? 16 : 0) + ((ln & 16) ? 8 : 0) + (e & 7);
        int Mg = mt * 16 + (ln & 15);
        sh.Af[idx] = (_Float16)sh.Av[Mg * DX + K];
    }
    __syncthreads();

    float t = ts[0];

    // ---------------- sequential Euler-Maruyama scan (4 barriers/step) ----------------
    #pragma unroll 1
    for (int s = 0; s < NSTEP; ++s) {
        const float dt   = ts[s + 1] - ts[s];
        const float sq   = SQRTF(dt);
        const float rinv = RCPF(T - t + EPS_C);

        // P1: dW staging (all threads) + layer1 (all waves) + ra WMMA (waves 0-1)
        {
            const int idx2 = tid * 2;
            const int mb = idx2 >> 5, jb = idx2 & 31;
            const float2 dwv = *(const float2*)&dWs[((size_t)s * BB + b0 + mb) * DW + jb];
            v2h dp; dp[0] = (_Float16)(dwv.x * sq); dp[1] = (_Float16)(dwv.y * sq);
            *(v2h*)&sh.dwh[idx2] = dp;

            const int n    = lane & 15;
            const int rowb = wave * 16 + ((lane & 16) ? 8 : 0);
            v8f acc = {};
            v16h a = *(const v16h*)&sh.Wp1[(wave * 32 + lane) * 16];
            v16h b = load_b_act(sh.xh, DX, 0, lane);
            acc = wmma_f16(a, b, acc);
            v8h out;
            #pragma unroll
            for (int r = 0; r < 8; ++r)
                out[r] = (_Float16)TANHF(acc[r] + sh.b1v[rowb + r] + t * sh.W1r0[rowb + r]);
            *(v8h*)&sh.h1t[n * HH + rowb] = out;

            if (wave < 2) {                           // ra^T = a @ r^T (1 WMMA per wave)
                v8f racc = {};
                v16h aa = *(const v16h*)&sh.Af[(wave * 32 + lane) * 16];
                v16h bb = load_b_act(sh.rh, DX, 0, lane);
                racc = wmma_f16(aa, bb, racc);
                const int rowa = wave * 16 + ((lane & 16) ? 8 : 0);
                #pragma unroll
                for (int r = 0; r < 8; ++r) sh.ra[n * DX + rowa + r] = racc[r];
            }
        }
        __syncthreads();

        // P2: layer2 (all waves) + ll reduction (all threads, hides behind WMMAs)
        {
            const int n    = lane & 15;
            const int rowb = wave * 16 + ((lane & 16) ? 8 : 0);
            v8f acc = {};
            #pragma unroll
            for (int kt = 0; kt < 4; ++kt) {
                v16h a = *(const v16h*)&sh.Wp2[((wave * 4 + kt) * 32 + lane) * 16];
                v16h b = load_b_act(sh.h1t, HH, kt, lane);
                acc = wmma_f16(a, b, acc);
            }
            v8h out;
            #pragma unroll
            for (int r = 0; r < 8; ++r)
                out[r] = (_Float16)TANHF(acc[r] + sh.b2v[rowb + r]);
            *(v8h*)&sh.h2t[n * HH + rowb] = out;

            // Gval = f_base.r - 0.5 r.(r@a) ; 16-lane half-wave = one batch row
            const int idx2 = tid * 2;
            const int mb = idx2 >> 5, jb = idx2 & 31;
            const float xj0 = sh.x[idx2], xj1 = sh.x[idx2 + 1];
            const float r0  = (sh.vtv[jb]     - xj0) * rinv;
            const float r1  = (sh.vtv[jb + 1] - xj1) * rinv;
            float g = (BETA_C * (sh.muv[jb]     - xj0)) * r0 - 0.5f * r0 * sh.ra[idx2]
                    + (BETA_C * (sh.muv[jb + 1] - xj1)) * r1 - 0.5f * r1 * sh.ra[idx2 + 1];
            g += __shfl_xor(g, 1, 32);
            g += __shfl_xor(g, 2, 32);
            g += __shfl_xor(g, 4, 32);
            g += __shfl_xor(g, 8, 32);
            if ((lane & 15) == 0) sh.ll[mb] += g * dt;
        }
        __syncthreads();

        // P3: w0-1: v = W3^T@h2^T + b3 ; w2-3: vg = (W3G^T)^T@h2^T + b3g ; w4-5: wg = G@dW^T
        {
            const int n = lane & 15;
            if (wave < 2) {
                const int rowb = wave * 16 + ((lane & 16) ? 8 : 0);
                v8f acc = {};
                #pragma unroll
                for (int kt = 0; kt < 4; ++kt) {
                    v16h a = *(const v16h*)&sh.Wp3[((wave * 4 + kt) * 32 + lane) * 16];
                    v16h b = load_b_act(sh.h2t, HH, kt, lane);
                    acc = wmma_f16(a, b, acc);
                }
                #pragma unroll
                for (int r = 0; r < 8; ++r)
                    sh.vv[n * DW + rowb + r] = acc[r] + sh.b3v[rowb + r];
            } else if (wave < 4) {
                const int w = wave - 2;
                const int rowb = w * 16 + ((lane & 16) ? 8 : 0);
                v8f acc = {};
                #pragma unroll
                for (int kt = 0; kt < 4; ++kt) {
                    v16h a = *(const v16h*)&sh.W3Gf[((w * 4 + kt) * 32 + lane) * 16];
                    v16h b = load_b_act(sh.h2t, HH, kt, lane);
                    acc = wmma_f16(a, b, acc);
                }
                #pragma unroll
                for (int r = 0; r < 8; ++r)
                    sh.vgb[n * DX + rowb + r] = acc[r] + sh.b3g[rowb + r];
            } else if (wave < 6) {
                const int w = wave - 4;
                const int rowb = w * 16 + ((lane & 16) ? 8 : 0);
                v8f acc = {};
                v16h a = *(const v16h*)&sh.Gf[(w * 32 + lane) * 16];
                v16h b = load_b_act(sh.dwh, DW, 0, lane);
                acc = wmma_f16(a, b, acc);
                #pragma unroll
                for (int r = 0; r < 8; ++r)
                    sh.wgb[n * DX + rowb + r] = acc[r];
            }
        }
        __syncthreads();

        // P4: elementwise fp32 state update; emit xs/vs; produce next-step r^T f16
        {
            const int idx2 = tid * 2;
            const int mb = idx2 >> 5, i = idx2 & 31;
            const float rinvN = RCPF(T - (t + dt) + EPS_C);
            const float xv0 = sh.x[idx2], xv1 = sh.x[idx2 + 1];
            const float xn0 = xv0 + (BETA_C * (sh.muv[i]     - xv0) + sh.ra[idx2]     + sh.vgb[idx2])     * dt + sh.wgb[idx2];
            const float xn1 = xv1 + (BETA_C * (sh.muv[i + 1] - xv1) + sh.ra[idx2 + 1] + sh.vgb[idx2 + 1]) * dt + sh.wgb[idx2 + 1];
            sh.x[idx2]     = xn0;
            sh.x[idx2 + 1] = xn1;
            v2h xp; xp[0] = (_Float16)xn0; xp[1] = (_Float16)xn1;
            *(v2h*)&sh.xh[idx2] = xp;
            v2h rp; rp[0] = (_Float16)((sh.vtv[i]     - xn0) * rinvN);
                    rp[1] = (_Float16)((sh.vtv[i + 1] - xn1) * rinvN);
            *(v2h*)&sh.rh[idx2] = rp;
            const size_t o = ((size_t)s * BB + b0 + mb) * DX + i;
            *(float2*)&xs_o[o] = make_float2(xn0, xn1);
            *(float2*)&vs_o[o] = make_float2(sh.vv[idx2], sh.vv[idx2 + 1]);
        }
        t += dt;
        __syncthreads();
    }

    if (tid < BT) ll_o[b0 + tid] = sh.ll[tid];
}

extern "C" void kernel_launch(void* const* d_in, const int* in_sizes, int n_in,
                              void* d_out, int out_size, void* d_ws, size_t ws_size,
                              hipStream_t stream) {
    const float* x0  = (const float*)d_in[0];
    const float* ts  = (const float*)d_in[1];
    const float* dWs = (const float*)d_in[2];
    const float* W1  = (const float*)d_in[3];
    const float* b1  = (const float*)d_in[4];
    const float* W2  = (const float*)d_in[5];
    const float* b2  = (const float*)d_in[6];
    const float* W3  = (const float*)d_in[7];
    const float* b3  = (const float*)d_in[8];
    const float* Gm  = (const float*)d_in[9];
    const float* mu  = (const float*)d_in[10];
    const float* vt  = (const float*)d_in[11];

    float* xs = (float*)d_out;
    float* vs = xs + (size_t)NSTEP * BB * DX;
    float* ll = vs + (size_t)NSTEP * BB * DW;

    NeuralBridgeSDE_kernel<<<BB / BT, 256, 0, stream>>>(
        x0, ts, dWs, W1, b1, W2, b2, W3, b3, Gm, mu, vt, xs, vs, ll);
}
